// LSTM_42958262895380
// MI455X (gfx1250) — matrix operand order repaired and checked by reading, not compiled, not run on hardware
//
#include <hip/hip_runtime.h>
#include <hip/hip_bf16.h>

// ---------------- problem constants (from reference) ----------------
#define NN   65536
#define BSZ  64
#define INF_ 128
#define HID  256
#define OUTD 8
#define LAY  3
#define GMF  10
#define DECW 64

// padded LDS row stride for staged weight tiles: 256 halfs + 8 halfs pad
#define PADH 264              // halfs  (528 bytes = 33*16 -> 16B aligned rows)
#define TILE_ROWS 128         // 4 gates x 32 hidden cols
#define LDS_TILE_BYTES (TILE_ROWS * PADH * 2)   // 67584 B per matrix

typedef __attribute__((ext_vector_type(16))) __bf16        v16bf;
typedef __attribute__((ext_vector_type(8)))  float         v8f;
typedef __attribute__((ext_vector_type(4)))  float         f32x4;
typedef __attribute__((ext_vector_type(4)))  unsigned int  u32x4;
typedef __attribute__((ext_vector_type(4)))  unsigned int  tdm_u4;
typedef __attribute__((ext_vector_type(8)))  int           tdm_i8;
typedef __attribute__((ext_vector_type(4)))  int           tdm_i4;

// ---------------- WMMA helpers ----------------
__device__ __forceinline__ v8f wmma_bf16(v16bf a, v16bf b, v8f c) {
    return __builtin_amdgcn_wmma_f32_16x16x32_bf16(false, a, false, b,
                                                   (short)0, c, false, false);
}

// A fragment (16x32 bf16, M = lane&15): lanes 0-15 hold K sub..sub+7 and
// sub+16..sub+23 with sub=0; lanes 16-31 use sub=8.
__device__ __forceinline__ v16bf load_fragA_bf(const __bf16* rowptr, int kc, int lane) {
    const int sub = (lane & 16) ? 8 : 0;
    union U { u32x4 u[2]; v16bf v; } r;
    r.u[0] = *reinterpret_cast<const u32x4*>(rowptr + kc + sub);
    r.u[1] = *reinterpret_cast<const u32x4*>(rowptr + kc + sub + 16);
    return r.v;
}

// Same A fragment but source is f32; convert to bf16 on the fly.
__device__ __forceinline__ v16bf load_fragA_f32(const float* rowptr, int kc, int lane) {
    const int sub = (lane & 16) ? 8 : 0;
    const f32x4* p = reinterpret_cast<const f32x4*>(rowptr + kc + sub);
    const f32x4* q = reinterpret_cast<const f32x4*>(rowptr + kc + sub + 16);
    f32x4 c0 = p[0], c1 = p[1], c2 = q[0], c3 = q[1];
    v16bf r;
#pragma unroll
    for (int e = 0; e < 4; ++e) {
        r[e]      = (__bf16)c0[e];
        r[4 + e]  = (__bf16)c1[e];
        r[8 + e]  = (__bf16)c2[e];
        r[12 + e] = (__bf16)c3[e];
    }
    return r;
}

// B fragment (32x16 bf16, column n = lane&15): lanes 0-15 hold K 0..7 and
// 16..23, lanes 16-31 hold K 8..15 and 24..31 (same order as A).
__device__ __forceinline__ v16bf load_fragB(const __bf16* colptr, int kc, int lane) {
    const int sub = (lane & 16) ? 8 : 0;
    union U { u32x4 u[2]; v16bf v; } r;
    r.u[0] = *reinterpret_cast<const u32x4*>(colptr + kc + sub);
    r.u[1] = *reinterpret_cast<const u32x4*>(colptr + kc + sub + 16);
    return r.v;
}

__device__ __forceinline__ float sigmoidf_(float x) {
    return 1.0f / (1.0f + __expf(-x));
}

// ---------------- TDM: 2-D tile load Global -> LDS with row padding -------
// Loads tile_d1 rows x tile_d0 elems (2B each), row stride stride_d0 elems in
// memory; LDS destination gets 4 DWORDs (16B) of pad after every 128 DWORDs
// (512B = one 256-half row) -> padded row stride 528B.
__device__ __forceinline__ void tdm_load_tile_2d(unsigned lds_off,
                                                 const void* gptr,
                                                 unsigned tensor_d0,
                                                 unsigned tensor_d1,
                                                 unsigned tile_d0,
                                                 unsigned tile_d1,
                                                 unsigned stride_d0) {
    unsigned long long ga = (unsigned long long)(uintptr_t)gptr;
    tdm_u4 g0;
    g0[0] = 1u;                                        // count = 1 (valid)
    g0[1] = lds_off;                                   // LDS byte address
    g0[2] = (unsigned)(ga & 0xFFFFFFFFu);              // global addr lo
    g0[3] = (unsigned)((ga >> 32) & 0x1FFFFFFu)        // global addr hi
          | (2u << 30);                                // type = 2 ("image")
    tdm_i8 g1;
    g1[0] = (int)((1u << 16)      // data_size = 1 -> 2 bytes
                | (1u << 20)      // pad_enable
                | (6u << 22)      // pad_interval: 1<<6 8B-units = 128 DWORDs
                | (3u << 25));    // pad_amount: 4 DWORDs
    g1[1] = (int)((tensor_d0 & 0xFFFFu) << 16);
    g1[2] = (int)((tensor_d0 >> 16) | ((tensor_d1 & 0xFFFFu) << 16));
    g1[3] = (int)((tensor_d1 >> 16) | ((tile_d0 & 0xFFFFu) << 16));
    g1[4] = (int)(tile_d1 & 0xFFFFu);                  // tile_dim1, tile_dim2=0
    g1[5] = (int)stride_d0;                            // tensor_dim0_stride lo
    g1[6] = 0;
    g1[7] = 0;
    tdm_i4 z = {0, 0, 0, 0};
#if __clang_major__ >= 23
    tdm_i8 z8 = {0, 0, 0, 0, 0, 0, 0, 0};
    __builtin_amdgcn_tensor_load_to_lds(g0, g1, z, z, z8, 0);
#else
    __builtin_amdgcn_tensor_load_to_lds(g0, g1, z, z, 0);
#endif
}

// ---------------- prep kernels ----------------
__global__ void cvt_f32_bf16_kernel(const float* __restrict__ in,
                                    __bf16* __restrict__ out, int n) {
    int i = blockIdx.x * 256 + threadIdx.x;
    if (i < n) out[i] = (__bf16)in[i];
}

__global__ void pad_wd2_kernel(const float* __restrict__ wd2,
                               __bf16* __restrict__ out) {
    int r = blockIdx.x;          // 0..15
    int c = threadIdx.x;         // 0..63
    out[r * DECW + c] = (r < OUTD) ? (__bf16)wd2[r * DECW + c] : (__bf16)0.0f;
}

__global__ void prep_nodes_kernel(const float* __restrict__ gm,
                                  const float* __restrict__ gnode,
                                  const long long* __restrict__ ptr,
                                  __bf16* __restrict__ gm_bf,
                                  __bf16* __restrict__ x0) {
    const int n = blockIdx.x;
    const int k = threadIdx.x;   // 0..127
    int lo = 0, hi = BSZ;
    long long nn = (long long)n;
    while (lo < hi) {
        int mid = (lo + hi + 1) >> 1;
        if (nn >= ptr[mid]) lo = mid; else hi = mid - 1;
    }
    const int seg = lo;
    float v = (k < INF_ - GMF) ? gnode[n * INF_ + k]
                               : gm[seg * GMF + (k - (INF_ - GMF))];
    x0[n * INF_ + k] = (__bf16)v;
    if (k < 16)
        gm_bf[n * 16 + k] = (k < GMF) ? (__bf16)gm[seg * GMF + k] : (__bf16)0.0f;
}

// ---------------- encoder: X = relu(x0 @ Wenc^T + benc), bf16 out ----------
__global__ __launch_bounds__(256)
void encoder_kernel(const __bf16* __restrict__ x0,
                    const __bf16* __restrict__ Wenc,   // [256,128] bf16
                    const float* __restrict__ benc,
                    __bf16* __restrict__ X) {          // [N,256] bf16
    const int lane  = threadIdx.x & 31;
    const int wave  = threadIdx.x >> 5;
    const int rbase = blockIdx.x * 128 + wave * 16;
    const int jbase = blockIdx.y * 64;
    const int arow  = rbase + (lane & 15);
    const int bcol  = lane & 15;

    v8f acc[4] = { {}, {}, {}, {} };
    const __bf16* xrow = x0 + (size_t)arow * INF_;
    v16bf afrag[4];
#pragma unroll
    for (int k8 = 0; k8 < 4; ++k8) afrag[k8] = load_fragA_bf(xrow, k8 * 32, lane);
#pragma unroll
    for (int k8 = 0; k8 < 4; ++k8) {
#pragma unroll
        for (int s = 0; s < 4; ++s) {
            v16bf b = load_fragB(Wenc + (size_t)(jbase + s * 16 + bcol) * INF_,
                                 k8 * 32, lane);
            acc[s] = wmma_bf16(afrag[k8], b, acc[s]);
        }
    }
    const int rofs = (lane & 16) ? 8 : 0;
#pragma unroll
    for (int s = 0; s < 4; ++s) {
        const int j = jbase + s * 16 + (lane & 15);
        const float bj = benc[j];
#pragma unroll
        for (int v = 0; v < 8; ++v) {
            const int row = rbase + v + rofs;
            float h = acc[s][v] + bj;
            h = h > 0.0f ? h : 0.0f;
            X[(size_t)row * HID + j] = (__bf16)h;
        }
    }
}

// ---------------- fused LSTM layer ----------------
// gates = Xin @ Wih^T + H @ Whh^T + b ; elementwise cell update fused.
// Workgroup: 128 rows x 32 hidden cols (all four gates). Wave: 16 rows.
// Weight tiles are DMA'd into LDS by the Tensor Data Mover with padded rows.
__global__ __launch_bounds__(256)
void lstm_layer_kernel(const __bf16* __restrict__ Xin,    // [N,256] bf16
                       const float*  __restrict__ Hf,     // [N,256] f32
                       const float*  __restrict__ Cin,    // [N,256] f32
                       const __bf16* __restrict__ Wih,    // [1024,256] bf16
                       const __bf16* __restrict__ Whh,    // [1024,256] bf16
                       const float*  __restrict__ bih,    // [1024]
                       const float*  __restrict__ bhh,    // [1024]
                       const __bf16* __restrict__ gm_bf,  // [N,16] bf16
                       float* __restrict__ Hout,          // [N,256]
                       float* __restrict__ Cout,          // [N,256]
                       __bf16* __restrict__ Xnext,        // [N,256] or null
                       int write_xnext) {
    extern __shared__ char lds_raw[];
    __bf16* ldsWx = (__bf16*)lds_raw;                       // Wih tile (padded)
    __bf16* ldsWh = (__bf16*)(lds_raw + LDS_TILE_BYTES);    // Whh tile (padded)

    const int lane  = threadIdx.x & 31;
    const int wave  = threadIdx.x >> 5;
    const int rbase = blockIdx.x * 128 + wave * 16;
    const int jbase = blockIdx.y * 32;
    const int arow  = rbase + (lane & 15);

    // ---- stage both weight tiles into LDS via TDM (wave 0 issues) ----
    if (wave == 0) {
        const unsigned lds0 = (unsigned)(uintptr_t)ldsWx;
        const unsigned lds1 = (unsigned)(uintptr_t)ldsWh;
#pragma unroll
        for (int g = 0; g < 4; ++g) {
            const unsigned dofs = (unsigned)(g * 32 * PADH * 2);
            tdm_load_tile_2d(lds0 + dofs,
                             Wih + (size_t)(g * HID + jbase) * HID,
                             HID, 32, HID, 32, HID);
            tdm_load_tile_2d(lds1 + dofs,
                             Whh + (size_t)(g * HID + jbase) * HID,
                             HID, 32, HID, 32, HID);
        }
        __builtin_amdgcn_s_wait_tensorcnt(0);
    }
    __syncthreads();

    v8f acc[8];   // t = g*2+s
#pragma unroll
    for (int t = 0; t < 8; ++t) acc[t] = {};

    // per-tile LDS row pointer for this lane's B column
    const int brow = lane & 15;

    // ---- x-part: preload all A fragments, then 64 WMMAs from LDS ----
    {
        const __bf16* xrow = Xin + (size_t)arow * HID;
        v16bf afrag[8];
#pragma unroll
        for (int k8 = 0; k8 < 8; ++k8)
            afrag[k8] = load_fragA_bf(xrow, k8 * 32, lane);
#pragma unroll
        for (int k8 = 0; k8 < 8; ++k8) {
#pragma unroll
            for (int t = 0; t < 8; ++t) {
                const __bf16* bp = ldsWx + (size_t)(t * 16 + brow) * PADH;
                v16bf b = load_fragB(bp, k8 * 32, lane);
                acc[t] = wmma_bf16(afrag[k8], b, acc[t]);
            }
        }
    }
    // ---- h-part (f32 source, converted on the fly) ----
    {
        const float* hrow = Hf + (size_t)arow * HID;
        v16bf afrag[8];
#pragma unroll
        for (int k8 = 0; k8 < 8; ++k8)
            afrag[k8] = load_fragA_f32(hrow, k8 * 32, lane);
#pragma unroll
        for (int k8 = 0; k8 < 8; ++k8) {
#pragma unroll
            for (int t = 0; t < 8; ++t) {
                const __bf16* bp = ldsWh + (size_t)(t * 16 + brow) * PADH;
                v16bf b = load_fragB(bp, k8 * 32, lane);
                acc[t] = wmma_bf16(afrag[k8], b, acc[t]);
            }
        }
    }

    // ---- fused LSTM elementwise epilogue ----
    const int rofs = (lane & 16) ? 8 : 0;
#pragma unroll
    for (int s = 0; s < 2; ++s) {
        const int j = jbase + s * 16 + (lane & 15);
        const float bi = bih[0 * HID + j] + bhh[0 * HID + j];
        const float bf = bih[1 * HID + j] + bhh[1 * HID + j];
        const float bg = bih[2 * HID + j] + bhh[2 * HID + j];
        const float bo = bih[3 * HID + j] + bhh[3 * HID + j];
#pragma unroll
        for (int v = 0; v < 8; ++v) {
            const int row = rbase + v + rofs;
            const float iv = sigmoidf_(acc[0 * 2 + s][v] + bi);
            const float fv = sigmoidf_(acc[1 * 2 + s][v] + bf);
            const float gv = tanhf(acc[2 * 2 + s][v] + bg);
            const float ov = sigmoidf_(acc[3 * 2 + s][v] + bo);
            const float c_old = Cin[(size_t)row * HID + j];
            const float cn = fv * c_old + iv * gv;
            const float hn = ov * tanhf(cn);
            Hout[(size_t)row * HID + j] = hn;
            Cout[(size_t)row * HID + j] = cn;
            if (write_xnext) {
                float xv = (j < HID - GMF) ? hn
                         : (float)gm_bf[(size_t)row * 16 + (j - (HID - GMF))];
                Xnext[(size_t)row * HID + j] = (__bf16)xv;
            }
        }
    }
}

// ---------------- decoder ----------------
__global__ __launch_bounds__(256)
void decoder_kernel(const float* __restrict__ Hs2,    // [N,256] f32
                    const float* __restrict__ Cs2,    // [N,256] f32
                    const __bf16* __restrict__ Wd1,   // [64,512] bf16
                    const float* __restrict__ bd1,    // [64]
                    const __bf16* __restrict__ Wd2p,  // [16,64] bf16 padded
                    const float* __restrict__ bd2,    // [8]
                    float* __restrict__ y) {          // [N,8]
    __shared__ __bf16 hidlds[8][16][DECW];
    const int lane  = threadIdx.x & 31;
    const int wave  = threadIdx.x >> 5;
    const int rbase = blockIdx.x * 128 + wave * 16;
    const int arow  = rbase + (lane & 15);
    const int bcol  = lane & 15;

    v8f acc[4] = { {}, {}, {}, {} };
    const float* hrow = Hs2 + (size_t)arow * HID;
    const float* crow = Cs2 + (size_t)arow * HID;
    for (int kc = 0; kc < 2 * HID; kc += 32) {
        const float* src = (kc < HID) ? hrow : crow;
        v16bf a = load_fragA_f32(src, kc & (HID - 1), lane);
#pragma unroll
        for (int s = 0; s < 4; ++s) {
            v16bf b = load_fragB(Wd1 + (size_t)(s * 16 + bcol) * (2 * HID), kc, lane);
            acc[s] = wmma_bf16(a, b, acc[s]);
        }
    }
    const int rofs = (lane & 16) ? 8 : 0;
#pragma unroll
    for (int s = 0; s < 4; ++s) {
        const int c = s * 16 + (lane & 15);
        const float bj = bd1[c];
#pragma unroll
        for (int v = 0; v < 8; ++v) {
            float h = acc[s][v] + bj;
            h = h > 0.0f ? h : 0.0f;
            hidlds[wave][v + rofs][c] = (__bf16)h;
        }
    }
    __syncthreads();

    v8f acc2 = {};
    const __bf16* lrow = &hidlds[wave][lane & 15][0];
#pragma unroll
    for (int kc = 0; kc < DECW; kc += 32) {
        v16bf a = load_fragA_bf(lrow, kc, lane);
        v16bf b = load_fragB(Wd2p + (size_t)bcol * DECW, kc, lane);
        acc2 = wmma_bf16(a, b, acc2);
    }
    const int col = lane & 15;
    if (col < OUTD) {
#pragma unroll
        for (int v = 0; v < 8; ++v) {
            const int row = rbase + v + rofs;
            float yy = acc2[v] + bd2[col];
            y[(size_t)row * OUTD + col] = yy > 0.0f ? yy : 0.0f;
        }
    }
}

// ---------------- host launcher ----------------
extern "C" void kernel_launch(void* const* d_in, const int* in_sizes, int n_in,
                              void* d_out, int out_size, void* d_ws, size_t ws_size,
                              hipStream_t stream) {
    const float*      gm    = (const float*)d_in[0];
    const float*      gnode = (const float*)d_in[1];
    const long long*  ptr   = (const long long*)d_in[4];
    const float*      H     = (const float*)d_in[5];
    const float*      C     = (const float*)d_in[6];
    const float*      Wenc  = (const float*)d_in[7];
    const float*      benc  = (const float*)d_in[8];
    const float*      Wih   = (const float*)d_in[9];
    const float*      Whh   = (const float*)d_in[10];
    const float*      bih   = (const float*)d_in[11];
    const float*      bhh   = (const float*)d_in[12];
    const float*      Wd1   = (const float*)d_in[13];
    const float*      bd1   = (const float*)d_in[14];
    const float*      Wd2   = (const float*)d_in[15];
    const float*      bd2   = (const float*)d_in[16];

    float* out = (float*)d_out;
    float* Hs  = out;                              // [3,N,256]
    float* Cs  = out + (size_t)LAY * NN * HID;     // [3,N,256]
    float* y   = out + (size_t)2 * LAY * NN * HID; // [N,8]

    char* base = (char*)d_ws;
    size_t off = 0;
    auto carve = [&](size_t bytes) {
        void* p = base + off;
        off += (bytes + 255) & ~(size_t)255;
        return p;
    };
    __bf16* gm_bf  = (__bf16*)carve((size_t)NN * 16 * 2);
    __bf16* x0     = (__bf16*)carve((size_t)NN * INF_ * 2);
    __bf16* Xa     = (__bf16*)carve((size_t)NN * HID * 2);
    __bf16* Xb     = (__bf16*)carve((size_t)NN * HID * 2);
    __bf16* Wencbf = (__bf16*)carve((size_t)HID * INF_ * 2);
    __bf16* Wihbf  = (__bf16*)carve((size_t)LAY * 4 * HID * HID * 2);
    __bf16* Whhbf  = (__bf16*)carve((size_t)LAY * 4 * HID * HID * 2);
    __bf16* Wd1bf  = (__bf16*)carve((size_t)DECW * 2 * HID * 2);
    __bf16* Wd2p   = (__bf16*)carve((size_t)16 * DECW * 2);

    {
        int n;
        n = HID * INF_;
        cvt_f32_bf16_kernel<<<(n + 255) / 256, 256, 0, stream>>>(Wenc, Wencbf, n);
        n = LAY * 4 * HID * HID;
        cvt_f32_bf16_kernel<<<(n + 255) / 256, 256, 0, stream>>>(Wih, Wihbf, n);
        cvt_f32_bf16_kernel<<<(n + 255) / 256, 256, 0, stream>>>(Whh, Whhbf, n);
        n = DECW * 2 * HID;
        cvt_f32_bf16_kernel<<<(n + 255) / 256, 256, 0, stream>>>(Wd1, Wd1bf, n);
        pad_wd2_kernel<<<16, DECW, 0, stream>>>(Wd2, Wd2p);
    }

    prep_nodes_kernel<<<NN, INF_, 0, stream>>>(gm, gnode, ptr, gm_bf, x0);

    encoder_kernel<<<dim3(NN / 128, HID / 64), 256, 0, stream>>>(x0, Wencbf, benc, Xa);

    const __bf16* xin[LAY]  = { Xa, Xb, Xa };
    __bf16*       xnxt[LAY] = { Xb, Xa, nullptr };
    const size_t lds_bytes = 2 * LDS_TILE_BYTES;   // 132 KB: two padded tiles
    for (int l = 0; l < LAY; ++l) {
        lstm_layer_kernel<<<dim3(NN / 128, HID / 32), 256, lds_bytes, stream>>>(
            xin[l],
            H + (size_t)l * NN * HID,
            C + (size_t)l * NN * HID,
            Wihbf + (size_t)l * 4 * HID * HID,
            Whhbf + (size_t)l * 4 * HID * HID,
            bih + (size_t)l * 4 * HID,
            bhh + (size_t)l * 4 * HID,
            gm_bf,
            Hs + (size_t)l * NN * HID,
            Cs + (size_t)l * NN * HID,
            xnxt[l],
            (l < LAY - 1) ? 1 : 0);
    }

    decoder_kernel<<<NN / 128, 256, 0, stream>>>(
        Hs + (size_t)(LAY - 1) * NN * HID,
        Cs + (size_t)(LAY - 1) * NN * HID,
        Wd1bf, bd1, Wd2p, bd2, y);
}
